// scMGCN_56882546868392
// MI455X (gfx1250) — matrix operand order repaired and checked by reading, not compiled, not run on hardware
//
#include <hip/hip_runtime.h>
#include <math.h>

#define N_NODES 50000
#define E_EDGES 1600000
#define G_GRAPHS 3
#define IN_DIM 2000
#define H_DIM 64
#define OUT_DIM 16
#define HATT_DIM 128
#define KT32 63                 /* 2016 / 32 */
#define K_PAD (KT32 * 32)
#define MT16 (N_NODES / 16)     /* 3125 m-tiles, exact */
#define NT16 (H_DIM / 16)       /* 4 n-tiles */

typedef __attribute__((ext_vector_type(16))) _Float16 v16h;
typedef __attribute__((ext_vector_type(8)))  float    v8f;
typedef __attribute__((ext_vector_type(2)))  float    v2f;

// ---------------------------------------------------- integer degrees ---
__global__ void deg_kernel(const int* __restrict__ src, const int* __restrict__ dst,
                           int* __restrict__ degs, int* __restrict__ degd) {
  int g = blockIdx.y;
  int e = blockIdx.x * blockDim.x + threadIdx.x;
  if (e >= E_EDGES) return;
  int s = src[(long)g * E_EDGES + e];
  int d = dst[(long)g * E_EDGES + e];
  atomicAdd(&degs[(long)g * N_NODES + s], 1);
  atomicAdd(&degd[(long)g * N_NODES + d], 1);
}

__global__ void norm_fin_kernel(const int* __restrict__ deg, float* __restrict__ nrm) {
  long t = (long)blockIdx.x * blockDim.x + threadIdx.x;
  if (t >= (long)G_GRAPHS * N_NODES) return;
  nrm[t] = rsqrtf(fmaxf((float)deg[t], 1.0f));
}

// --------------------- exclusive prefix scan of dst-degrees -> rowptr ---
__global__ void scan_kernel(const int* __restrict__ deg, int* __restrict__ rowptr) {
  int g = blockIdx.x;
  __shared__ int sdata[1024];
  __shared__ int running;
  if (threadIdx.x == 0) running = 0;
  __syncthreads();
  for (int base0 = 0; base0 < N_NODES; base0 += 1024) {
    int i = base0 + threadIdx.x;
    int v = (i < N_NODES) ? deg[(long)g * N_NODES + i] : 0;
    sdata[threadIdx.x] = v;
    __syncthreads();
    for (int offc = 1; offc < 1024; offc <<= 1) {
      int t = (threadIdx.x >= offc) ? sdata[threadIdx.x - offc] : 0;
      __syncthreads();
      sdata[threadIdx.x] += t;
      __syncthreads();
    }
    int incl = sdata[threadIdx.x];
    int excl = incl - v;
    int runs = running;
    if (i < N_NODES) rowptr[(long)g * (N_NODES + 1) + i] = runs + excl;
    __syncthreads();
    if (threadIdx.x == 1023) running = runs + incl;
    __syncthreads();
  }
  if (threadIdx.x == 0) rowptr[(long)g * (N_NODES + 1) + N_NODES] = running;
}

// --------------------------------------- fill CSR buckets with src ids ---
__global__ void fill_kernel(const int* __restrict__ src, const int* __restrict__ dst,
                            const int* __restrict__ rowptr, int* __restrict__ cursor,
                            int* __restrict__ csr_src) {
  int g = blockIdx.y;
  int e = blockIdx.x * blockDim.x + threadIdx.x;
  if (e >= E_EDGES) return;
  int s = src[(long)g * E_EDGES + e];
  int d = dst[(long)g * E_EDGES + e];
  int pos = atomicAdd(&cursor[(long)g * N_NODES + d], 1);
  csr_src[(long)g * E_EDGES + rowptr[(long)g * (N_NODES + 1) + d] + pos] = s;
}

// --------------------------------------------- pack x -> f16 A fragments ---
__global__ void pack_x_kernel(const float* __restrict__ x, unsigned short* __restrict__ xb) {
  long tid = (long)blockIdx.x * blockDim.x + threadIdx.x;
  if (tid >= (long)MT16 * KT32 * 32) return;
  int lane = (int)(tid & 31);
  long r = tid >> 5;
  int kt = (int)(r % KT32);
  int mt = (int)(r / KT32);
  int m = mt * 16 + (lane & 15);
  int half = lane >> 4;
  _Float16* outp = (_Float16*)(xb + (((long)mt * KT32 + kt) * 32 + lane) * 16);
#pragma unroll
  for (int e = 0; e < 16; ++e) {
    int v = e >> 1, p = e & 1;
    int koff = (v < 4) ? (half * 8 + 2 * v + p) : (16 + half * 8 + 2 * (v - 4) + p);
    int k = kt * 32 + koff;
    float val = (k < IN_DIM) ? x[(long)m * IN_DIM + k] : 0.0f;
    outp[e] = (_Float16)val;
  }
}

// ------------------------------------------- pack W0 -> f16 B fragments ---
__global__ void pack_w_kernel(const float* __restrict__ W0, unsigned short* __restrict__ wb) {
  int tid = blockIdx.x * blockDim.x + threadIdx.x;
  if (tid >= G_GRAPHS * KT32 * NT16 * 32) return;
  int lane = tid & 31;
  int r = tid >> 5;
  int nt = r % NT16; r /= NT16;
  int kt = r % KT32;
  int g  = r / KT32;
  int half = lane >> 4;
  int n = nt * 16 + (lane & 15);
  _Float16* outp = (_Float16*)(wb + ((((long)g * KT32 + kt) * NT16 + nt) * 32 + lane) * 16);
#pragma unroll
  for (int e = 0; e < 16; ++e) {
    int k = kt * 32 + ((e < 8) ? (half * 8 + e) : (16 + half * 8 + (e - 8)));
    float val = (k < IN_DIM) ? W0[((long)g * IN_DIM + k) * H_DIM + n] : 0.0f;
    outp[e] = (_Float16)val;
  }
}

// --------------------- big GEMM: h = (x @ W0[g]) * ns[row]  (f16 WMMA) ---
__global__ void gemm_x_w0(const unsigned short* __restrict__ xb,
                          const unsigned short* __restrict__ wb,
                          const float* __restrict__ ns,
                          float* __restrict__ h) {
  int wave = blockIdx.x * (blockDim.x >> 5) + (threadIdx.x >> 5);
  if (wave >= G_GRAPHS * MT16) return;
  int lane = threadIdx.x & 31;
  int g  = wave / MT16;
  int mt = wave % MT16;
  v8f acc0 = {}, acc1 = {}, acc2 = {}, acc3 = {};
  for (int kt = 0; kt < KT32; ++kt) {
    v16h a = *(const v16h*)(xb + (((long)mt * KT32 + kt) * 32 + lane) * 16);
    const unsigned short* wbase = wb + (((long)g * KT32 + kt) * NT16) * 32 * 16;
    v16h b0 = *(const v16h*)(wbase + ((long)(0 * 32 + lane)) * 16);
    v16h b1 = *(const v16h*)(wbase + ((long)(1 * 32 + lane)) * 16);
    v16h b2 = *(const v16h*)(wbase + ((long)(2 * 32 + lane)) * 16);
    v16h b3 = *(const v16h*)(wbase + ((long)(3 * 32 + lane)) * 16);
    acc0 = __builtin_amdgcn_wmma_f32_16x16x32_f16(false, a, false, b0, (short)0, acc0, false, false);
    acc1 = __builtin_amdgcn_wmma_f32_16x16x32_f16(false, a, false, b1, (short)0, acc1, false, false);
    acc2 = __builtin_amdgcn_wmma_f32_16x16x32_f16(false, a, false, b2, (short)0, acc2, false, false);
    acc3 = __builtin_amdgcn_wmma_f32_16x16x32_f16(false, a, false, b3, (short)0, acc3, false, false);
  }
  int half = lane >> 4;
  int col  = lane & 15;
  float* hb = h + ((long)g * N_NODES + (long)mt * 16) * H_DIM;
  const float* nsb = ns + (long)g * N_NODES + (long)mt * 16;
#pragma unroll
  for (int v = 0; v < 8; ++v) {
    int m = v + 8 * half;
    float w = nsb[m];
    hb[(long)m * H_DIM +  0 + col] = acc0[v] * w;
    hb[(long)m * H_DIM + 16 + col] = acc1[v] * w;
    hb[(long)m * H_DIM + 32 + col] = acc2[v] * w;
    hb[(long)m * H_DIM + 48 + col] = acc3[v] * w;
  }
}

// ---------------- atomic-free CSR gather: out[n] = sum_in hw[src] (L2) ---
__global__ void gather_kernel(const float* __restrict__ hw, const int* __restrict__ rowptr,
                              const int* __restrict__ csr_src, float* __restrict__ outp) {
  int g = blockIdx.y;
  long t = (long)blockIdx.x * blockDim.x + threadIdx.x;
  if (t >= (long)N_NODES * 16) return;
  int q = (int)(t & 15);
  int n = (int)(t >> 4);
  int beg = rowptr[(long)g * (N_NODES + 1) + n];
  int end = rowptr[(long)g * (N_NODES + 1) + n + 1];
  const int* cs = csr_src + (long)g * E_EDGES;
  const float* hg = hw + (long)g * N_NODES * H_DIM + q * 4;
  float4 acc = make_float4(0.f, 0.f, 0.f, 0.f);
  for (int i = beg; i < end; ++i) {
    int s = cs[i];
    float4 v = *(const float4*)(hg + (long)s * H_DIM);
    acc.x += v.x; acc.y += v.y; acc.z += v.z; acc.w += v.w;
  }
  *(float4*)(outp + ((long)g * N_NODES + n) * H_DIM + q * 4) = acc;
}

// ------------------- layer-0 epilogue: elu(acc*nd + b0) * ns -> out ---
__global__ void pointwise0_kernel(const float* __restrict__ acc, const float* __restrict__ nd,
                                  const float* __restrict__ ns, const float* __restrict__ b0,
                                  float* __restrict__ outp) {
  long t = (long)blockIdx.x * blockDim.x + threadIdx.x;
  if (t >= (long)G_GRAPHS * N_NODES * H_DIM) return;
  int c = (int)(t % H_DIM);
  long gn = t / H_DIM;
  int g = (int)(gn / N_NODES);
  int n = (int)(gn % N_NODES);
  float v = acc[t] * nd[(long)g * N_NODES + n] + b0[g * H_DIM + c];
  v = v > 0.0f ? v : (expf(v) - 1.0f);
  outp[t] = v * ns[(long)g * N_NODES + n];
}

// ---- small f32 GEMM: out = post((S*nd) @ W[g] + b[g]) via 16x16x4 WMMA ----
// post_mode 1: out = elu(x) * ns[row]   (hidden layers);  0: out = x (final z)
__global__ void gemm_small(const float* __restrict__ S, const float* __restrict__ nd,
                           const float* __restrict__ ns,
                           const float* __restrict__ W, const float* __restrict__ bias,
                           float* __restrict__ outp, int post_mode) {
  int wave = blockIdx.x * (blockDim.x >> 5) + (threadIdx.x >> 5);
  if (wave >= G_GRAPHS * MT16) return;
  int lane = threadIdx.x & 31;
  int g  = wave / MT16;
  int mt = wave % MT16;
  int half = lane >> 4;
  int lrow = lane & 15;
  int row = mt * 16 + lrow;
  float ndv = nd[(long)g * N_NODES + row];
  const float* srow = S + ((long)g * N_NODES + row) * H_DIM;
  const float* wg   = W + (long)g * H_DIM * H_DIM;
  v8f acc0 = {}, acc1 = {}, acc2 = {}, acc3 = {};
  for (int kb = 0; kb < H_DIM; kb += 4) {
    int k0 = kb + half * 2;
    v2f a;  a.x = srow[k0] * ndv;  a.y = srow[k0 + 1] * ndv;
    v2f b0v, b1v, b2v, b3v;
    b0v.x = wg[k0 * H_DIM +  0 + lrow];  b0v.y = wg[(k0 + 1) * H_DIM +  0 + lrow];
    b1v.x = wg[k0 * H_DIM + 16 + lrow];  b1v.y = wg[(k0 + 1) * H_DIM + 16 + lrow];
    b2v.x = wg[k0 * H_DIM + 32 + lrow];  b2v.y = wg[(k0 + 1) * H_DIM + 32 + lrow];
    b3v.x = wg[k0 * H_DIM + 48 + lrow];  b3v.y = wg[(k0 + 1) * H_DIM + 48 + lrow];
    acc0 = __builtin_amdgcn_wmma_f32_16x16x4_f32(false, a, false, b0v, (short)0, acc0, false, false);
    acc1 = __builtin_amdgcn_wmma_f32_16x16x4_f32(false, a, false, b1v, (short)0, acc1, false, false);
    acc2 = __builtin_amdgcn_wmma_f32_16x16x4_f32(false, a, false, b2v, (short)0, acc2, false, false);
    acc3 = __builtin_amdgcn_wmma_f32_16x16x4_f32(false, a, false, b3v, (short)0, acc3, false, false);
  }
  float* ob = outp + ((long)g * N_NODES + (long)mt * 16) * H_DIM;
  const float* bg = bias + g * H_DIM;
  const float* nsb = ns + (long)g * N_NODES + (long)mt * 16;
#pragma unroll
  for (int v = 0; v < 8; ++v) {
    int m = v + 8 * half;
    float x0 = acc0[v] + bg[ 0 + lrow];
    float x1 = acc1[v] + bg[16 + lrow];
    float x2 = acc2[v] + bg[32 + lrow];
    float x3 = acc3[v] + bg[48 + lrow];
    if (post_mode == 1) {
      float w = nsb[m];
      x0 = (x0 > 0.0f ? x0 : (expf(x0) - 1.0f)) * w;
      x1 = (x1 > 0.0f ? x1 : (expf(x1) - 1.0f)) * w;
      x2 = (x2 > 0.0f ? x2 : (expf(x2) - 1.0f)) * w;
      x3 = (x3 > 0.0f ? x3 : (expf(x3) - 1.0f)) * w;
    }
    ob[(long)m * H_DIM +  0 + lrow] = x0;
    ob[(long)m * H_DIM + 16 + lrow] = x1;
    ob[(long)m * H_DIM + 32 + lrow] = x2;
    ob[(long)m * H_DIM + 48 + lrow] = x3;
  }
}

// --------------------------- attention scores: wsum[g] += tanh(z@A1)@A2 ---
__global__ void attn_kernel(const float* __restrict__ z, const float* __restrict__ A1,
                            const float* __restrict__ a1b, const float* __restrict__ A2,
                            float* __restrict__ wsum) {
  int g = blockIdx.y;
  int n = blockIdx.x * blockDim.x + threadIdx.x;
  float s = 0.0f;
  if (n < N_NODES) {
    const float* zr = z + ((long)g * N_NODES + n) * H_DIM;
    float zv[H_DIM];
#pragma unroll
    for (int c = 0; c < H_DIM; ++c) zv[c] = zr[c];
    for (int j = 0; j < HATT_DIM; ++j) {
      float acc = a1b[j];
#pragma unroll 16
      for (int c = 0; c < H_DIM; ++c) acc += zv[c] * A1[c * HATT_DIM + j];
      s += tanhf(acc) * A2[j];
    }
  }
  for (int off = 16; off > 0; off >>= 1) s += __shfl_down(s, off, 32);
  if ((threadIdx.x & 31) == 0) atomicAdd(&wsum[g], s);
}

__global__ void beta_kernel(const float* __restrict__ wsum, float* __restrict__ beta) {
  if (threadIdx.x == 0 && blockIdx.x == 0) {
    float w0 = wsum[0] / (float)N_NODES;
    float w1 = wsum[1] / (float)N_NODES;
    float w2 = wsum[2] / (float)N_NODES;
    float m = fmaxf(w0, fmaxf(w1, w2));
    float e0 = expf(w0 - m), e1 = expf(w1 - m), e2 = expf(w2 - m);
    float s = e0 + e1 + e2;
    beta[0] = e0 / s; beta[1] = e1 / s; beta[2] = e2 / s;
  }
}

// ------------------- fuse + project + row-normalize -> d_out (N x 16) ---
__global__ void final_kernel(const float* __restrict__ z, const float* __restrict__ beta,
                             const float* __restrict__ Wp, const float* __restrict__ bp,
                             float* __restrict__ outp) {
  int n = blockIdx.x * blockDim.x + threadIdx.x;
  if (n >= N_NODES) return;
  float btm0 = beta[0], btm1 = beta[1], btm2 = beta[2];
  float fused[H_DIM];
#pragma unroll
  for (int c = 0; c < H_DIM; ++c) {
    fused[c] = btm0 * z[((long)0 * N_NODES + n) * H_DIM + c]
             + btm1 * z[((long)1 * N_NODES + n) * H_DIM + c]
             + btm2 * z[((long)2 * N_NODES + n) * H_DIM + c];
  }
  float logit[OUT_DIM];
  float nrm = 0.0f;
#pragma unroll
  for (int o = 0; o < OUT_DIM; ++o) {
    float acc = bp[o];
#pragma unroll 16
    for (int c = 0; c < H_DIM; ++c) acc += fused[c] * Wp[c * OUT_DIM + o];
    logit[o] = acc;
    nrm += acc * acc;
  }
  nrm = sqrtf(nrm);
  float inv = 1.0f / fmaxf(nrm, 1e-12f);
#pragma unroll
  for (int o = 0; o < OUT_DIM; ++o) outp[(long)n * OUT_DIM + o] = logit[o] * inv;
}

// =============================================================== launch ===
extern "C" void kernel_launch(void* const* d_in, const int* in_sizes, int n_in,
                              void* d_out, int out_size, void* d_ws, size_t ws_size,
                              hipStream_t stream) {
  const float* x   = (const float*)d_in[0];
  const int*   src = (const int*)  d_in[1];
  const int*   dst = (const int*)  d_in[2];
  const float* W0  = (const float*)d_in[3];
  const float* b0  = (const float*)d_in[4];
  const float* W1  = (const float*)d_in[5];
  const float* b1  = (const float*)d_in[6];
  const float* W2  = (const float*)d_in[7];
  const float* b2  = (const float*)d_in[8];
  const float* A1  = (const float*)d_in[9];
  const float* a1b = (const float*)d_in[10];
  const float* A2  = (const float*)d_in[11];
  const float* Wp  = (const float*)d_in[12];
  const float* bp  = (const float*)d_in[13];
  float* outp = (float*)d_out;

  char* base = (char*)d_ws;
  size_t off = 0;
  auto carve = [&](size_t bytes) -> void* {
    void* p = base + off;
    off += (bytes + 255) & ~(size_t)255;
    return p;
  };
  unsigned short* xb = (unsigned short*)carve((size_t)N_NODES * K_PAD * 2);
  unsigned short* wb = (unsigned short*)carve((size_t)G_GRAPHS * K_PAD * H_DIM * 2);
  int*   degs_i = (int*)  carve((size_t)G_GRAPHS * N_NODES * 4);
  int*   degd_i = (int*)  carve((size_t)G_GRAPHS * N_NODES * 4);
  float* ns     = (float*)carve((size_t)G_GRAPHS * N_NODES * 4);
  float* nd     = (float*)carve((size_t)G_GRAPHS * N_NODES * 4);
  int*   rowptr = (int*)  carve((size_t)G_GRAPHS * (N_NODES + 1) * 4);
  int*   cursor = (int*)  carve((size_t)G_GRAPHS * N_NODES * 4);
  int*   csrsrc = (int*)  carve((size_t)G_GRAPHS * E_EDGES * 4);
  float* bufA   = (float*)carve((size_t)G_GRAPHS * N_NODES * H_DIM * 4);
  float* bufB   = (float*)carve((size_t)G_GRAPHS * N_NODES * H_DIM * 4);
  float* bufC   = (float*)carve((size_t)G_GRAPHS * N_NODES * H_DIM * 4);
  float* wsum   = (float*)carve(256);
  float* beta   = (float*)carve(256);

  // --- degrees, norms, CSR build (topology fixed within the call) ---
  hipMemsetAsync(degs_i, 0, (size_t)G_GRAPHS * N_NODES * 4, stream);
  hipMemsetAsync(degd_i, 0, (size_t)G_GRAPHS * N_NODES * 4, stream);
  hipMemsetAsync(cursor, 0, (size_t)G_GRAPHS * N_NODES * 4, stream);
  hipMemsetAsync(wsum,   0, 256, stream);
  deg_kernel<<<dim3(E_EDGES / 256, G_GRAPHS), 256, 0, stream>>>(src, dst, degs_i, degd_i);
  {
    int blocks = (G_GRAPHS * N_NODES + 255) / 256;
    norm_fin_kernel<<<blocks, 256, 0, stream>>>(degs_i, ns);
    norm_fin_kernel<<<blocks, 256, 0, stream>>>(degd_i, nd);
  }
  scan_kernel<<<G_GRAPHS, 1024, 0, stream>>>(degd_i, rowptr);
  fill_kernel<<<dim3(E_EDGES / 256, G_GRAPHS), 256, 0, stream>>>(src, dst, rowptr, cursor, csrsrc);

  // --- pack x / W0 into f16 WMMA fragments ---
  {
    long tx = (long)MT16 * KT32 * 32;
    pack_x_kernel<<<(int)((tx + 255) / 256), 256, 0, stream>>>(x, xb);
    int tw = G_GRAPHS * KT32 * NT16 * 32;
    pack_w_kernel<<<(tw + 255) / 256, 256, 0, stream>>>(W0, wb);
  }

  const int gemm_blocks = (G_GRAPHS * MT16 + 7) / 8;
  const dim3 gather_grid(N_NODES * 16 / 256, G_GRAPHS);
  const int pw_blocks = (int)(((long)G_GRAPHS * N_NODES * H_DIM + 255) / 256);

  // --- layer 0: GEMM (x@W0)*ns -> gather -> elu(*nd+b)*ns ---
  gemm_x_w0<<<gemm_blocks, 256, 0, stream>>>(xb, wb, ns, bufA);
  gather_kernel<<<gather_grid, 256, 0, stream>>>(bufA, rowptr, csrsrc, bufB);
  pointwise0_kernel<<<pw_blocks, 256, 0, stream>>>(bufB, nd, ns, b0, bufA);

  // --- layer 1: gather -> ((S*nd)@W1+b1), elu, *ns ---
  gather_kernel<<<gather_grid, 256, 0, stream>>>(bufA, rowptr, csrsrc, bufB);
  gemm_small<<<gemm_blocks, 256, 0, stream>>>(bufB, nd, ns, W1, b1, bufA, 1);

  // --- layer 2: gather -> ((S*nd)@W2+b2) -> z ---
  gather_kernel<<<gather_grid, 256, 0, stream>>>(bufA, rowptr, csrsrc, bufB);
  gemm_small<<<gemm_blocks, 256, 0, stream>>>(bufB, nd, ns, W2, b2, bufC, 0);

  // --- attention, softmax, fuse+project+normalize ---
  attn_kernel<<<dim3((N_NODES + 127) / 128, G_GRAPHS), 128, 0, stream>>>(bufC, A1, a1b, A2, wsum);
  beta_kernel<<<1, 1, 0, stream>>>(wsum, beta);
  final_kernel<<<(N_NODES + 255) / 256, 256, 0, stream>>>(bufC, beta, Wp, bp, outp);

  (void)in_sizes; (void)n_in; (void)out_size; (void)ws_size;
}
